// Attention_8323646620215
// MI455X (gfx1250) — matrix operand order repaired and checked by reading, not compiled, not run on hardware
//
#include <hip/hip_runtime.h>

#define BATCH   4
#define NSEQ    2048
#define DMODEL  512
#define HEADS   8
#define DHEAD   64
#define MAXREL  200

typedef __attribute__((ext_vector_type(16))) __bf16 v16bf;
typedef __attribute__((ext_vector_type(8)))  __bf16 v8bf;
typedef __attribute__((ext_vector_type(8)))  float  v8f;
typedef __attribute__((ext_vector_type(4)))  unsigned int v4u;
typedef __attribute__((ext_vector_type(8)))  int v8i;
typedef __attribute__((ext_vector_type(4)))  int v4i;

union F8     { v8f   v; float  f[8];  };
union BF16x16{ v16bf v; __bf16 e[16]; };

__device__ __forceinline__ v8f wmma_bf16(v16bf a, v16bf b, v8f c) {
  return __builtin_amdgcn_wmma_f32_16x16x32_bf16(
      false, a, false, b, (short)0, c, false, false);
}

// A-fragment (16x32, 16-bit): lane<16 -> M=lane, K=0..7 & 16..23; lane>=16 -> M=lane-16, K=8..15 & 24..31
__device__ __forceinline__ v16bf load_a_frag(const __bf16* base, int stride, int k0, int lane) {
  const int hp = lane >> 4, lx = lane & 15;
  const __bf16* p = base + (size_t)lx * stride + k0 + hp * 8;
  v8bf lo = *(const v8bf*)p;
  v8bf hi = *(const v8bf*)(p + 16);
  return __builtin_shufflevector(lo, hi, 0,1,2,3,4,5,6,7,8,9,10,11,12,13,14,15);
}

// B-fragment (32x16, 16-bit): lane<16 -> N=lane, K=0..7 & 16..23; lane>=16 -> N=lane-16, K=8..15 & 24..31
__device__ __forceinline__ v16bf load_b_frag(const __bf16* base, int stride, int k0, int lane) {
  const int hp = lane >> 4, lx = lane & 15;
  const __bf16* p = base + (size_t)lx * stride + k0 + hp * 8;
  v8bf lo = *(const v8bf*)p;
  v8bf hi = *(const v8bf*)(p + 16);
  return __builtin_shufflevector(lo, hi, 0,1,2,3,4,5,6,7,8,9,10,11,12,13,14,15);
}

// ---- Tensor Data Mover: build D# (ISA cdna5 ch.8) and issue tensor_load_to_lds ----
// data_size = 8B elements. 1D copy: tile_d1 = 0. 2D: tile rows of tile_d0 elems,
// tile_d1 rows, row stride = stride0 (8B elems).
__device__ __forceinline__ void tdm_load(unsigned lds_off, const void* gptr,
                                         unsigned tile_d0, unsigned tile_d1,
                                         unsigned long long stride0) {
  const unsigned long long ga = (unsigned long long)gptr;
  v4u g0 = { 1u,                                   // count=1, is_restore=0, gather=0
             lds_off,                               // lds_addr (bytes)
             (unsigned)ga,                          // global_addr[31:0]
             (unsigned)((ga >> 32) & 0x01ffffffu) | (2u << 30) }; // ga[56:32] | type=2
  const unsigned dim0 = 0x00100000u;                // generous tensor dims (no OOB clipping)
  const unsigned dim1 = 0x00100000u;
  const unsigned w0 = (3u << 16);                   // data_size=3 (8B), mask=0, no pad/iter
  const unsigned w1 = (dim0 & 0xffffu) << 16;       // tensor_dim0[15:0] @ bits 63:48
  const unsigned w2 = ((dim0 >> 16) & 0xffffu) | ((dim1 & 0xffffu) << 16);
  const unsigned w3 = ((dim1 >> 16) & 0xffffu) | ((tile_d0 & 0xffffu) << 16);
  const unsigned w4 = (tile_d1 & 0xffffu);          // tile_dim1 (0 = 1 row)
  const unsigned w5 = (unsigned)(stride0 & 0xffffffffu);         // tensor_dim0_stride lo
  const unsigned w6 = (unsigned)((stride0 >> 32) & 0xffffu);     // tensor_dim0_stride hi
  v8i g1 = { (int)w0, (int)w1, (int)w2, (int)w3, (int)w4, (int)w5, (int)w6, 0 };
  v4i gz = { 0, 0, 0, 0 };
#if __clang_major__ >= 23
  v8i gz8 = { 0, 0, 0, 0, 0, 0, 0, 0 };
  __builtin_amdgcn_tensor_load_to_lds(g0, g1, gz, gz, gz8, 0);
#else
  __builtin_amdgcn_tensor_load_to_lds(g0, g1, gz, gz, 0);
#endif
}

// ---------------- Kernel 1: LayerNorm -> bf16 ----------------
__global__ void __launch_bounds__(256) ln_kernel(const float* __restrict__ x,
                                                 const float* __restrict__ gamma,
                                                 const float* __restrict__ beta,
                                                 __bf16* __restrict__ xn) {
  const int row = blockIdx.x;
  const int t   = threadIdx.x;
  const float* xr = x + (size_t)row * DMODEL;
  float v0 = xr[t], v1 = xr[t + 256];
  __shared__ float red[256];
  red[t] = v0 + v1; __syncthreads();
  for (int o = 128; o > 0; o >>= 1) { if (t < o) red[t] += red[t + o]; __syncthreads(); }
  const float mu = red[0] * (1.0f / DMODEL);
  __syncthreads();
  const float d0 = v0 - mu, d1 = v1 - mu;
  red[t] = d0 * d0 + d1 * d1; __syncthreads();
  for (int o = 128; o > 0; o >>= 1) { if (t < o) red[t] += red[t + o]; __syncthreads(); }
  const float rstd = rsqrtf(red[0] * (1.0f / DMODEL) + 1e-5f);
  __bf16* orow = xn + (size_t)row * DMODEL;
  orow[t]       = (__bf16)(d0 * rstd * gamma[t]       + beta[t]);
  orow[t + 256] = (__bf16)(d1 * rstd * gamma[t + 256] + beta[t + 256]);
}

// ---------------- Kernel 2: weight convert + transpose -> bf16 ----------------
__global__ void __launch_bounds__(256) wconv_kernel(const float* __restrict__ Wqkv,
                                                    const float* __restrict__ Wout,
                                                    __bf16* __restrict__ wq_t,
                                                    __bf16* __restrict__ wo_t) {
  const int idx = blockIdx.x * blockDim.x + threadIdx.x;
  if (idx < DMODEL * 3 * DMODEL) {
    const int k = idx / (3 * DMODEL), n = idx % (3 * DMODEL);
    wq_t[(size_t)n * DMODEL + k] = (__bf16)Wqkv[idx];
  }
  if (idx < DMODEL * DMODEL) {
    const int k = idx / DMODEL, n = idx % DMODEL;
    wo_t[(size_t)n * DMODEL + k] = (__bf16)Wout[idx];
  }
}

// ---------------- Kernel 3: QKV GEMM, 64x64 register tile per wave ----------------
__global__ void __launch_bounds__(256) qkv_gemm_kernel(const __bf16* __restrict__ xn,
                                                       const __bf16* __restrict__ wq_t,
                                                       __bf16* __restrict__ qb,   // [B*H][N][64]
                                                       __bf16* __restrict__ kb,   // [B*H][N][64]
                                                       __bf16* __restrict__ vt) { // [B*H][64][N]
  const int wid  = (blockIdx.x * blockDim.x + threadIdx.x) >> 5;
  const int lane = threadIdx.x & 31;
  const int TN = (3 * DMODEL) / 64;                 // 24 column tiles of 64
  const int tm = wid / TN, tn = wid % TN;           // tm in [0,128)
  const int m0 = tm * 64, n0 = tn * 64;
  F8 acc[4][4];
  for (int i = 0; i < 4; ++i) for (int j = 0; j < 4; ++j)
    for (int r = 0; r < 8; ++r) acc[i][j].f[r] = 0.f;
#pragma unroll 2
  for (int k0 = 0; k0 < DMODEL; k0 += 32) {
    v16bf a[4], b[4];
#pragma unroll
    for (int i = 0; i < 4; ++i)
      a[i] = load_a_frag(xn + (size_t)(m0 + 16 * i) * DMODEL, DMODEL, k0, lane);
#pragma unroll
    for (int j = 0; j < 4; ++j)
      b[j] = load_b_frag(wq_t + (size_t)(n0 + 16 * j) * DMODEL, DMODEL, k0, lane);
#pragma unroll
    for (int i = 0; i < 4; ++i)
#pragma unroll
      for (int j = 0; j < 4; ++j)
        acc[i][j].v = wmma_bf16(a[i], b[j], acc[i][j].v);
  }
  const int hp = lane >> 4, lx = lane & 15;
  for (int j = 0; j < 4; ++j) {
    const int cn  = n0 + j * 16 + lx;
    const int seg = cn / DMODEL;                    // uniform per j-tile (64 | 512)
    const int c   = cn % DMODEL;
    const int h   = c / DHEAD, d = c % DHEAD;
    for (int i = 0; i < 4; ++i) {
      for (int r = 0; r < 8; ++r) {
        const int rm = m0 + i * 16 + r + 8 * hp;
        const int b_ = rm / NSEQ, n = rm % NSEQ;
        const size_t bh = (size_t)b_ * HEADS + h;
        const float val = acc[i][j].f[r];
        if (seg == 0)      qb[(bh * NSEQ + n) * DHEAD + d] = (__bf16)(val * 0.125f);
        else if (seg == 1) kb[(bh * NSEQ + n) * DHEAD + d] = (__bf16)val;
        else               vt[(bh * DHEAD + d) * NSEQ + n] = (__bf16)val;
      }
    }
  }
}

// ---------------- Kernel 4: flash attention, TDM double-buffered K/V in LDS ----------------
__global__ void __launch_bounds__(256) attn_kernel(const __bf16* __restrict__ qb,
                                                   const __bf16* __restrict__ kb,
                                                   const __bf16* __restrict__ vt,
                                                   const float* __restrict__ rel_table,
                                                   const int* __restrict__ tmask,
                                                   __bf16* __restrict__ ao) {
  __shared__ __bf16 kst[2][32 * DHEAD];             // K chunk: 32 keys x 64d  (4 KB each)
  __shared__ __bf16 vst[2][DHEAD * 32];             // V chunk: 64d x 32 keys  (4 KB each)
  __shared__ float  Pl[8][16][32];                  // per-wave P bounce tile  (16 KB)
  const int wwid = threadIdx.x >> 5;
  const int wid  = blockIdx.x * 8 + wwid;
  const int lane = threadIdx.x & 31;
  const int hp = lane >> 4, lx = lane & 15;
  const int QT = NSEQ / 16;                         // 128 -> all 8 waves share bh
  const int bh = wid / QT, qt = wid % QT;
  const int n0 = qt * 16;
  const __bf16* qrow = qb + ((size_t)bh * NSEQ + n0) * DHEAD;
  const __bf16* kbh  = kb + (size_t)bh * NSEQ * DHEAD;
  const __bf16* vbh  = vt + (size_t)bh * DHEAD * NSEQ;

  const v16bf aq0 = load_a_frag(qrow, DHEAD, 0,  lane);
  const v16bf aq1 = load_a_frag(qrow, DHEAD, 32, lane);

  F8 acc[4];
  for (int dt = 0; dt < 4; ++dt) for (int r = 0; r < 8; ++r) acc[dt].f[r] = 0.f;
  float mrow[8], lrow[8];
  for (int r = 0; r < 8; ++r) { mrow[r] = -3.0e38f; lrow[r] = 0.f; }

  float (*P)[32] = Pl[wwid];

  // prime the pipeline: chunk 0 -> buffer 0 (TDM, one wave issues for whole block)
  if (wwid == 0) {
    tdm_load((unsigned)(unsigned long long)&kst[0][0], kbh, 512, 0, 0);          // 4 KB linear
    tdm_load((unsigned)(unsigned long long)&vst[0][0], vbh, 8, DHEAD, 512);      // 64 rows x 64 B
  }

  const int NCHUNK = NSEQ / 32;                     // 64
  for (int ci = 0; ci < NCHUNK; ++ci) {
    const int cur = ci & 1;
    const int mb  = ci * 32;
    if (wwid == 0) {
      if (ci + 1 < NCHUNK) {                        // kick next chunk, then wait for current
        tdm_load((unsigned)(unsigned long long)&kst[1 - cur][0],
                 kbh + (size_t)(mb + 32) * DHEAD, 512, 0, 0);
        tdm_load((unsigned)(unsigned long long)&vst[1 - cur][0],
                 vbh + (mb + 32), 8, DHEAD, 512);
        __builtin_amdgcn_s_wait_tensorcnt(2);
      } else {
        __builtin_amdgcn_s_wait_tensorcnt(0);
      }
    }
    __syncthreads();                                // K/V chunk `cur` visible to all waves

    // ---- scores: S[16q][32k], K-depth = 64, operands from LDS ----
    F8 s0, s1;
    for (int r = 0; r < 8; ++r) { s0.f[r] = 0.f; s1.f[r] = 0.f; }
    const __bf16* kt = &kst[cur][0];
    s0.v = wmma_bf16(aq0, load_b_frag(kt,              DHEAD, 0,  lane), s0.v);
    s0.v = wmma_bf16(aq1, load_b_frag(kt,              DHEAD, 32, lane), s0.v);
    s1.v = wmma_bf16(aq0, load_b_frag(kt + 16 * DHEAD, DHEAD, 0,  lane), s1.v);
    s1.v = wmma_bf16(aq1, load_b_frag(kt + 16 * DHEAD, DHEAD, 32, lane), s1.v);

    // ---- rel-pos bias + temporal mask + row max ----
    const int m0c = mb + lx, m1c = mb + 16 + lx;
    float pv0[8], pv1[8], cmax[8];
    for (int r = 0; r < 8; ++r) {
      const int nq = n0 + r + 8 * hp;
      int rel0 = min(max(nq - m0c, -(MAXREL - 1)), MAXREL - 1) + (MAXREL - 1);
      int rel1 = min(max(nq - m1c, -(MAXREL - 1)), MAXREL - 1) + (MAXREL - 1);
      float v0 = s0.f[r] + rel_table[rel0];
      float v1 = s1.f[r] + rel_table[rel1];
      if (tmask[(size_t)nq * NSEQ + m0c] == 0) v0 = -1e9f;
      if (tmask[(size_t)nq * NSEQ + m1c] == 0) v1 = -1e9f;
      pv0[r] = v0; pv1[r] = v1;
      float mx = fmaxf(v0, v1);
      mx = fmaxf(mx, __shfl_xor(mx, 1, 32));
      mx = fmaxf(mx, __shfl_xor(mx, 2, 32));
      mx = fmaxf(mx, __shfl_xor(mx, 4, 32));
      mx = fmaxf(mx, __shfl_xor(mx, 8, 32));
      cmax[r] = mx;
    }
    // ---- online softmax; stage P (C layout) in LDS ----
    float alpha[8];
    for (int r = 0; r < 8; ++r) {
      const float newm = fmaxf(mrow[r], cmax[r]);
      alpha[r] = __expf(mrow[r] - newm);
      mrow[r] = newm;
      const float e0 = __expf(pv0[r] - newm);
      const float e1 = __expf(pv1[r] - newm);
      float ss = e0 + e1;
      ss += __shfl_xor(ss, 1, 32);
      ss += __shfl_xor(ss, 2, 32);
      ss += __shfl_xor(ss, 4, 32);
      ss += __shfl_xor(ss, 8, 32);
      lrow[r] = lrow[r] * alpha[r] + ss;
      const int rr = r + 8 * hp;
      P[rr][lx]      = e0;
      P[rr][16 + lx] = e1;
    }
    for (int dt = 0; dt < 4; ++dt)
      for (int r = 0; r < 8; ++r) acc[dt].f[r] *= alpha[r];

    asm volatile("s_wait_dscnt 0x0" ::: "memory");  // per-wave in-order DS: P is readable

    // ---- transpose P into A-fragment layout (bf16) ----
    BF16x16 ap;
    const int kb0 = hp * 8;
    for (int i = 0; i < 8; ++i) {
      ap.e[i]     = (__bf16)P[lx][kb0 + i];
      ap.e[8 + i] = (__bf16)P[lx][16 + kb0 + i];
    }
    // ---- acc += P(16x32) @ V(32x64), V from LDS ----
#pragma unroll
    for (int dt = 0; dt < 4; ++dt)
      acc[dt].v = wmma_bf16(ap.v, load_b_frag(&vst[cur][dt * 16 * 32], 32, 0, lane), acc[dt].v);

    __syncthreads();                                // done reading buffers before next overwrite
  }

  const int b_ = bh / HEADS, h = bh % HEADS;
  for (int dt = 0; dt < 4; ++dt) {
    for (int r = 0; r < 8; ++r) {
      const int nq = n0 + r + 8 * hp;
      const int dcol = dt * 16 + lx;
      const float o = acc[dt].f[r] / lrow[r];
      ao[((size_t)b_ * NSEQ + nq) * DMODEL + h * DHEAD + dcol] = (__bf16)o;
    }
  }
}

// ---------------- Kernel 5: output GEMM + bias, 64x64 tile per wave ----------------
__global__ void __launch_bounds__(256) out_gemm_kernel(const __bf16* __restrict__ ao,
                                                       const __bf16* __restrict__ wo_t,
                                                       const float* __restrict__ bout,
                                                       float* __restrict__ out) {
  const int wid  = (blockIdx.x * blockDim.x + threadIdx.x) >> 5;
  const int lane = threadIdx.x & 31;
  const int TN = DMODEL / 64;                       // 8
  const int tm = wid / TN, tn = wid % TN;
  const int m0 = tm * 64, n0 = tn * 64;
  F8 acc[4][4];
  for (int i = 0; i < 4; ++i) for (int j = 0; j < 4; ++j)
    for (int r = 0; r < 8; ++r) acc[i][j].f[r] = 0.f;
#pragma unroll 2
  for (int k0 = 0; k0 < DMODEL; k0 += 32) {
    v16bf a[4], b[4];
#pragma unroll
    for (int i = 0; i < 4; ++i)
      a[i] = load_a_frag(ao + (size_t)(m0 + 16 * i) * DMODEL, DMODEL, k0, lane);
#pragma unroll
    for (int j = 0; j < 4; ++j)
      b[j] = load_b_frag(wo_t + (size_t)(n0 + 16 * j) * DMODEL, DMODEL, k0, lane);
#pragma unroll
    for (int i = 0; i < 4; ++i)
#pragma unroll
      for (int j = 0; j < 4; ++j)
        acc[i][j].v = wmma_bf16(a[i], b[j], acc[i][j].v);
  }
  const int hp = lane >> 4, lx = lane & 15;
  for (int j = 0; j < 4; ++j) {
    const int cn = n0 + j * 16 + lx;
    const float bia = bout[cn];
    for (int i = 0; i < 4; ++i)
      for (int r = 0; r < 8; ++r) {
        const int rm = m0 + i * 16 + r + 8 * hp;
        out[(size_t)rm * DMODEL + cn] = acc[i][j].f[r] + bia;
      }
  }
}

extern "C" void kernel_launch(void* const* d_in, const int* in_sizes, int n_in,
                              void* d_out, int out_size, void* d_ws, size_t ws_size,
                              hipStream_t stream) {
  (void)in_sizes; (void)n_in; (void)out_size; (void)ws_size;
  const float* x         = (const float*)d_in[0];
  const float* gamma     = (const float*)d_in[1];
  const float* beta      = (const float*)d_in[2];
  const float* Wqkv      = (const float*)d_in[3];
  const float* Wout      = (const float*)d_in[4];
  const float* bout      = (const float*)d_in[5];
  const float* rel_table = (const float*)d_in[6];
  const int*   tmask     = (const int*)d_in[7];
  float* out = (float*)d_out;

  size_t off = 0;
  auto take = [&](size_t bytes) -> void* {
    void* p = (char*)d_ws + off;
    off += (bytes + 255) & ~(size_t)255;
    return p;
  };
  __bf16* xn  = (__bf16*)take((size_t)BATCH * NSEQ * DMODEL * 2);
  __bf16* wqt = (__bf16*)take((size_t)3 * DMODEL * DMODEL * 2);
  __bf16* wot = (__bf16*)take((size_t)DMODEL * DMODEL * 2);
  __bf16* qb  = (__bf16*)take((size_t)BATCH * HEADS * NSEQ * DHEAD * 2);
  __bf16* kb  = (__bf16*)take((size_t)BATCH * HEADS * NSEQ * DHEAD * 2);
  __bf16* vt  = (__bf16*)take((size_t)BATCH * HEADS * NSEQ * DHEAD * 2);
  __bf16* ao  = (__bf16*)take((size_t)BATCH * NSEQ * DMODEL * 2);

  ln_kernel<<<BATCH * NSEQ, 256, 0, stream>>>(x, gamma, beta, xn);
  wconv_kernel<<<(DMODEL * 3 * DMODEL) / 256, 256, 0, stream>>>(Wqkv, Wout, wqt, wot);
  { // (8192/64) x (1536/64) = 128 x 24 = 3072 waves
    const int waves = (BATCH * NSEQ / 64) * ((3 * DMODEL) / 64);
    qkv_gemm_kernel<<<waves / 8, 256, 0, stream>>>(xn, wqt, qb, kb, vt);
  }
  { // B*H * N/16 = 4096 waves, 8 per block sharing one (b,h) K/V stream
    const int waves = BATCH * HEADS * (NSEQ / 16);
    attn_kernel<<<waves / 8, 256, 0, stream>>>(qb, kb, vt, rel_table, tmask, ao);
  }
  { // (8192/64) x (512/64) = 128 x 8 = 1024 waves
    const int waves = (BATCH * NSEQ / 64) * (DMODEL / 64);
    out_gemm_kernel<<<waves / 8, 256, 0, stream>>>(ao, wot, bout, out);
  }
}
